// UnifiedQCNNQLSTM_65481071407854
// MI455X (gfx1250) — matrix operand order repaired and checked by reading, not compiled, not run on hardware
//
#include <hip/hip_runtime.h>

// ---------------- types ----------------
typedef __attribute__((ext_vector_type(16))) __bf16 v16bf;
typedef __attribute__((ext_vector_type(8)))  __bf16 v8bf;
typedef __attribute__((ext_vector_type(8)))  float  v8f;

#define SEQ   16384
#define HDIM  2048
#define TAG   2048
#define LSTM_WGS 64
#define LSTM_THREADS 256
#define KSPLIT 8
#define KCH   256     // HDIM / KSPLIT

// float -> bf16 (round to nearest even)
__device__ __forceinline__ unsigned short f2bf(float f) {
    unsigned u = __float_as_uint(f);
    unsigned r = u + 0x7FFFu + ((u >> 16) & 1u);
    return (unsigned short)(r >> 16);
}

__device__ __forceinline__ float sigmoidf_(float x) {
    return 1.0f / (1.0f + expf(-x));
}

// ---------------- 1) QCNN per-token MLP ----------------
__global__ __launch_bounds__(256) void qcnn_kernel(
    const float* __restrict__ sent,
    const float* __restrict__ fm_w, const float* __restrict__ fm_b,
    const float* __restrict__ c1_w, const float* __restrict__ c1_b,
    const float* __restrict__ p1_w, const float* __restrict__ p1_b,
    const float* __restrict__ c2_w, const float* __restrict__ c2_b,
    const float* __restrict__ p2_w, const float* __restrict__ p2_b,
    const float* __restrict__ c3_w, const float* __restrict__ c3_b,
    const float* __restrict__ h_w,  const float* __restrict__ h_b,
    float* __restrict__ xseq)
{
    int t = blockIdx.x * 256 + threadIdx.x;
    if (t >= SEQ) return;
    const float* x = sent + (size_t)t * 8;
    float a[16], b[16];
    #pragma unroll
    for (int o = 0; o < 16; ++o) { float s = fm_b[o];
        #pragma unroll
        for (int i = 0; i < 8;  ++i) s = fmaf(x[i], fm_w[i*16+o], s); a[o] = tanhf(s); }
    #pragma unroll
    for (int o = 0; o < 16; ++o) { float s = c1_b[o];
        #pragma unroll
        for (int i = 0; i < 16; ++i) s = fmaf(a[i], c1_w[i*16+o], s); b[o] = tanhf(s); }
    #pragma unroll
    for (int o = 0; o < 12; ++o) { float s = p1_b[o];
        #pragma unroll
        for (int i = 0; i < 16; ++i) s = fmaf(b[i], p1_w[i*12+o], s); a[o] = tanhf(s); }
    #pragma unroll
    for (int o = 0; o < 4;  ++o) { float s = c2_b[o];
        #pragma unroll
        for (int i = 0; i < 12; ++i) s = fmaf(a[i], c2_w[i*4+o], s);  b[o] = tanhf(s); }
    #pragma unroll
    for (int o = 0; o < 4;  ++o) { float s = p2_b[o];
        #pragma unroll
        for (int i = 0; i < 4;  ++i) s = fmaf(b[i], p2_w[i*4+o], s);  a[o] = tanhf(s); }
    #pragma unroll
    for (int o = 0; o < 4;  ++o) { float s = c3_b[o];
        #pragma unroll
        for (int i = 0; i < 4;  ++i) s = fmaf(a[i], c3_w[i*4+o], s);  b[o] = tanhf(s); }
    float s = h_b[0];
    #pragma unroll
    for (int i = 0; i < 4; ++i) s = fmaf(b[i], h_w[i], s);
    xseq[t] = sigmoidf_(s);
}

// ---------------- 2) pack wt (H x TAG f32) into WMMA-B bf16 tiles ----------------
// B lane layout (this code's k order): lane<16 -> N=lane, K=elem(0..15); lane>=16 -> N=lane-16, K=16+elem.
// Flat layout: [ntile][ktile][lane][elem], 16 bf16 contiguous per lane.
__global__ __launch_bounds__(256) void pack_wt_kernel(
    const float* __restrict__ wt, unsigned short* __restrict__ wtp)
{
    unsigned idx = blockIdx.x * 256u + threadIdx.x;   // < 128*64*32*16 = 4194304
    int e    = idx & 15;
    int lane = (idx >> 4) & 31;
    int kt   = (idx >> 9) & 63;
    int nt   = idx >> 15;
    int n = nt * 16 + (lane & 15);
    int k = kt * 32 + ((lane < 16) ? e : 16 + e);
    wtp[idx] = f2bf(wt[(size_t)k * TAG + n]);
}

// ---------------- 3) persistent LSTM ----------------
// 64 WGs x 256 threads. WG owns 32 hidden units. thread = (jg[3b], gate[2b], ks[3b]):
// each thread accumulates 4 consecutive gate columns (float4 weight loads) over a
// 256-row K-chunk; ks is wave-uniform so the LDS h read is a broadcast.
__global__ __launch_bounds__(LSTM_THREADS, 1) void lstm_kernel(
    const float* __restrict__ wf, const float* __restrict__ bfv,
    const float* __restrict__ wi, const float* __restrict__ biv,
    const float* __restrict__ wu, const float* __restrict__ buv,
    const float* __restrict__ wo, const float* __restrict__ bov,
    const float* __restrict__ xseq,
    float* __restrict__ hbuf,            // 2 * HDIM f32 (double buffer), zeroed at launch
    unsigned short* __restrict__ hseq,   // SEQ * HDIM bf16
    unsigned* __restrict__ cnt)          // zeroed at launch
{
    __shared__ float  hL[HDIM];
    __shared__ float4 part4[LSTM_THREADS];
    __shared__ float  cL[32];

    const int tid  = threadIdx.x;
    const int wg   = blockIdx.x;
    const int jg   = tid & 7;            // 8 column-groups of 4
    const int gate = (tid >> 3) & 3;
    const int ks   = tid >> 5;           // 0..7, uniform per wave
    const int j0   = wg * 32;
    const int jcol = j0 + jg * 4;

    const float* W  = (gate == 0) ? wf  : (gate == 1) ? wi  : (gate == 2) ? wu  : wo;
    const float* Bb = (gate == 0) ? bfv : (gate == 1) ? biv : (gate == 2) ? buv : bov;

    if (tid < 32) cL[tid] = 0.0f;

    float4* hL4 = reinterpret_cast<float4*>(hL);

    for (int s = 0; s < SEQ; ++s) {
        // stage previous h (full vector) into LDS, vectorized
        const float4* hprev4 =
            reinterpret_cast<const float4*>(hbuf + (size_t)(s & 1) * HDIM);
        for (int i = tid; i < HDIM / 4; i += LSTM_THREADS) hL4[i] = hprev4[i];
        __syncthreads();

        const int kbase = ks * KCH;
        const float* Wp = W + (size_t)(1 + kbase) * HDIM + jcol;
        float4 acc = make_float4(0.f, 0.f, 0.f, 0.f);
        #pragma unroll 4
        for (int t = 0; t < KCH; ++t) {
            const float  h = hL[kbase + t];
            const float4 w = *reinterpret_cast<const float4*>(Wp + (size_t)t * HDIM);
            acc.x = fmaf(h, w.x, acc.x);
            acc.y = fmaf(h, w.y, acc.y);
            acc.z = fmaf(h, w.z, acc.z);
            acc.w = fmaf(h, w.w, acc.w);
        }
        if (ks == 0) {
            const float  xv = xseq[s];
            const float4 w0 = *reinterpret_cast<const float4*>(W + jcol);
            const float4 bb = *reinterpret_cast<const float4*>(Bb + jcol);
            acc.x = fmaf(xv, w0.x, acc.x + bb.x);
            acc.y = fmaf(xv, w0.y, acc.y + bb.y);
            acc.z = fmaf(xv, w0.z, acc.z + bb.z);
            acc.w = fmaf(xv, w0.w, acc.w + bb.w);
        }
        part4[tid] = acc;
        __syncthreads();

        if (tid < 32) {
            const float* pf = reinterpret_cast<const float*>(part4);
            const int jg2 = tid >> 2, comp = tid & 3;
            float g[4];
            #pragma unroll
            for (int gg = 0; gg < 4; ++gg) {
                float sg = 0.0f;
                #pragma unroll
                for (int kk = 0; kk < KSPLIT; ++kk)
                    sg += pf[((kk * 32 + gg * 8 + jg2) << 2) + comp];
                g[gg] = sg;
            }
            float c = sigmoidf_(g[0]) * cL[tid] + sigmoidf_(g[1]) * tanhf(g[2]);
            cL[tid] = c;
            float h = sigmoidf_(g[3]) * tanhf(c);
            hbuf[(size_t)((s + 1) & 1) * HDIM + j0 + tid] = h;
            hseq[(size_t)s * HDIM + j0 + tid] = f2bf(h);
        }

        // grid-wide barrier (monotonic counter; reset by memset each launch)
        __syncthreads();
        if (tid == 0) {
            __threadfence();
            atomicAdd(cnt, 1u);
            const unsigned target = (unsigned)LSTM_WGS * (unsigned)(s + 1);
            volatile unsigned* vc = cnt;
            while (*vc < target) { __builtin_amdgcn_s_sleep(1); }
        }
        __syncthreads();
        __threadfence();
    }
}

// ---------------- 4) projection GEMM via WMMA bf16 ----------------
// out(M=16384, N=2048) = hseq(M,K bf16) @ wtp. Each wave: one 16-row strip x 4 N-tiles
// (A tile reused across 4 WMMAs per k-step).
__global__ __launch_bounds__(256) void gemm_wmma_kernel(
    const unsigned short* __restrict__ hseq,
    const unsigned short* __restrict__ wtp,
    float* __restrict__ out)
{
    const int lane = threadIdx.x & 31;
    const int wv   = threadIdx.x >> 5;
    const int task = blockIdx.x * 8 + wv;        // 32768 wave-tasks
    const int NTG  = TAG / 64;                   // 32 groups of 4 N-tiles
    const int KT   = HDIM / 32;                  // 64
    const int mt   = task / NTG;
    const int nt0  = (task % NTG) * 4;
    const int m0   = mt * 16;

    // A lane layout used here (contiguous k, same as B packing): lane<16 -> M=lane, K elems {0..15};
    // lane>=16 -> M=lane-16, K elems {16..31}
    const int mA    = m0 + (lane & 15);
    const int kOffA = (lane >> 4) * 16;          // 0 or 16

    v8f acc[4] = {};
    for (int kt = 0; kt < KT; ++kt) {
        const int k0 = kt * 32;
        const v8bf* ap =
            reinterpret_cast<const v8bf*>(hseq + (size_t)mA * HDIM + k0 + kOffA);
        v8bf alo = ap[0];     // K = k0+off .. +7
        v8bf ahi = ap[1];     // K = k0+off+8 .. +15
        v16bf a;
        #pragma unroll
        for (int e = 0; e < 8; ++e) { a[e] = alo[e]; a[8 + e] = ahi[e]; }
        #pragma unroll
        for (int i = 0; i < 4; ++i) {
            const v8bf* bp = reinterpret_cast<const v8bf*>(
                wtp + ((size_t)(((nt0 + i) * KT + kt) * 32 + lane) << 4));
            v8bf blo = bp[0];
            v8bf bhi = bp[1];
            v16bf b;
            #pragma unroll
            for (int e = 0; e < 8; ++e) { b[e] = blo[e]; b[8 + e] = bhi[e]; }
            acc[i] = __builtin_amdgcn_wmma_f32_16x16x32_bf16(
                false, a, false, b, (short)0, acc[i], false, false);
        }
    }
    // D layout: elem r -> M = m0 + r + 8*(lane>>4), N = n0 + (lane&15)
    const int nD0 = lane & 15;
    const int mB  = m0 + ((lane >> 4) << 3);
    #pragma unroll
    for (int i = 0; i < 4; ++i) {
        const int nD = (nt0 + i) * 16 + nD0;
        #pragma unroll
        for (int r = 0; r < 8; ++r)
            out[(size_t)(mB + r) * TAG + nD] = acc[i][r];
    }
}

// ---------------- 5) add bias + log_softmax, in place on d_out ----------------
__global__ __launch_bounds__(256) void lsm_kernel(
    float* __restrict__ out, const float* __restrict__ bt)
{
    __shared__ float red[256];
    float* o = out + (size_t)blockIdx.x * TAG;
    const int tid = threadIdx.x;

    float mx = -INFINITY;
    for (int c = tid; c < TAG; c += 256) mx = fmaxf(mx, o[c] + bt[c]);
    red[tid] = mx; __syncthreads();
    for (int s = 128; s > 0; s >>= 1) {
        if (tid < s) red[tid] = fmaxf(red[tid], red[tid + s]);
        __syncthreads();
    }
    mx = red[0]; __syncthreads();

    float sum = 0.0f;
    for (int c = tid; c < TAG; c += 256) sum += expf((o[c] + bt[c]) - mx);
    red[tid] = sum; __syncthreads();
    for (int s = 128; s > 0; s >>= 1) {
        if (tid < s) red[tid] += red[tid + s];
        __syncthreads();
    }
    float lse = mx + logf(red[0]);
    for (int c = tid; c < TAG; c += 256) o[c] = (o[c] + bt[c]) - lse;
}

// ---------------- launch ----------------
extern "C" void kernel_launch(void* const* d_in, const int* in_sizes, int n_in,
                              void* d_out, int out_size, void* d_ws, size_t ws_size,
                              hipStream_t stream) {
    const float* sent = (const float*)d_in[0];
    const float* fm_w = (const float*)d_in[1];  const float* fm_b = (const float*)d_in[2];
    const float* c1_w = (const float*)d_in[3];  const float* c1_b = (const float*)d_in[4];
    const float* p1_w = (const float*)d_in[5];  const float* p1_b = (const float*)d_in[6];
    const float* c2_w = (const float*)d_in[7];  const float* c2_b = (const float*)d_in[8];
    const float* p2_w = (const float*)d_in[9];  const float* p2_b = (const float*)d_in[10];
    const float* c3_w = (const float*)d_in[11]; const float* c3_b = (const float*)d_in[12];
    const float* h_w  = (const float*)d_in[13]; const float* h_b  = (const float*)d_in[14];
    const float* wf   = (const float*)d_in[15]; const float* bfv  = (const float*)d_in[16];
    const float* wi   = (const float*)d_in[17]; const float* biv  = (const float*)d_in[18];
    const float* wu   = (const float*)d_in[19]; const float* buv  = (const float*)d_in[20];
    const float* wo   = (const float*)d_in[21]; const float* bov  = (const float*)d_in[22];
    const float* wt   = (const float*)d_in[23]; const float* bt   = (const float*)d_in[24];
    float* out = (float*)d_out;

    char* ws = (char*)d_ws;
    unsigned*       cnt  = (unsigned*)ws;                         // [0, 4)
    float*          hbuf = (float*)(ws + 256);                    // 2*2048 f32
    float*          xseq = (float*)(ws + 32768);                  // 16384 f32
    unsigned short* wtp  = (unsigned short*)(ws + 131072);        // 8 MB
    unsigned short* hseq = (unsigned short*)(ws + (16ull << 20)); // 64 MB

    // reset barrier counter + h double buffer (init h0 = 0) each launch
    hipMemsetAsync(d_ws, 0, 32768, stream);

    qcnn_kernel<<<SEQ / 256, 256, 0, stream>>>(
        sent, fm_w, fm_b, c1_w, c1_b, p1_w, p1_b, c2_w, c2_b,
        p2_w, p2_b, c3_w, c3_b, h_w, h_b, xseq);

    pack_wt_kernel<<<(HDIM * TAG) / 256, 256, 0, stream>>>(wt, wtp);

    lstm_kernel<<<LSTM_WGS, LSTM_THREADS, 0, stream>>>(
        wf, bfv, wi, biv, wu, buv, wo, bov, xseq, hbuf, hseq, cnt);

    gemm_wmma_kernel<<<(SEQ / 16) * (TAG / 64) / 8, 256, 0, stream>>>(hseq, wtp, out);

    lsm_kernel<<<SEQ, 256, 0, stream>>>(out, bt);
}
